// TrajDecoder_46050639348128
// MI455X (gfx1250) — matrix operand order repaired and checked by reading, not compiled, not run on hardware
//
#include <hip/hip_runtime.h>
#include <hip/hip_bf16.h>

#define DIM 512
#define KROWS 65536
#define TILE_ROWS 64

typedef __attribute__((ext_vector_type(16))) __bf16 v16bf;
typedef __attribute__((ext_vector_type(8)))  float v8f;
typedef __attribute__((ext_vector_type(8)))  unsigned short u16x8;
typedef __attribute__((ext_vector_type(4)))  unsigned short u16x4;

union Frag16 {
  v16bf v;
  u16x8 h[2];
  unsigned short s[16];
};

// Hardware fp32 -> bf16 (RNE) via the backend's cvt path.
__device__ __forceinline__ unsigned short f2bf(float f) {
  union { __bf16 b; unsigned short s; } u;
  u.b = (__bf16)f;
  return u.s;
}
__device__ __forceinline__ float bf2f(unsigned short b) {
  return __uint_as_float(((unsigned int)b) << 16);
}

// One-shot fp32 -> bf16 weight conversion (weights are L2-resident; 8MB->4MB).
__global__ void __launch_bounds__(256)
cvt_w_kernel(const float* __restrict__ src, unsigned short* __restrict__ dst, int n4) {
  const int i = blockIdx.x * blockDim.x + threadIdx.x;
  if (i < n4) {
    const float4 f = ((const float4*)src)[i];
    u16x4 o;
    o[0] = f2bf(f.x); o[1] = f2bf(f.y); o[2] = f2bf(f.z); o[3] = f2bf(f.w);
    ((u16x4*)dst)[i] = o;
  }
}

// One fused MLP layer for a 64-row tile:
//   dst[64x512](bf16) = relu(src[64x512](bf16) @ W^T + bias)
// W is bf16 row-major [h=512][c=512] (pre-converted, L2 resident).
// 8 waves; each computes 4 row-tiles x 4 col-tiles of 16x16 with
// v_wmma_f32_16x16x32_bf16 (fp32 accumulate). Inner loop is pure
// ds_load_b128 + global_load_b128 + wmma -- no conversion VALU.
__device__ __forceinline__ void fused_layer(
    const unsigned short* __restrict__ srcLds,
    unsigned short* __restrict__ dstLds,
    const unsigned short* __restrict__ Wb,   // bf16 weights
    const float* __restrict__ bias,
    int wave, int lane)
{
  const int m15 = lane & 15;
  const int hi  = lane >> 4;   // 0 or 1 (lane group)

  v8f acc[16];
  v8f vzero = {0.f, 0.f, 0.f, 0.f, 0.f, 0.f, 0.f, 0.f};
#pragma unroll
  for (int t = 0; t < 16; ++t) acc[t] = vzero;

  for (int kc = 0; kc < DIM; kc += 32) {
    // A fragments: 16-bit A 16x32 layout.
    // lanes 0-15: row M=lane, halves 0..7 -> K=kc+0..7, halves 8..15 -> K=kc+16..23
    // lanes 16-31: row M=lane-16, K offsets shifted by +8
    Frag16 a[4];
#pragma unroll
    for (int rt = 0; rt < 4; ++rt) {
      const int base = (rt * 16 + m15) * DIM + kc + hi * 8;
      a[rt].h[0] = *(const u16x8*)(srcLds + base);
      a[rt].h[1] = *(const u16x8*)(srcLds + base + 16);
    }
#pragma unroll
    for (int j = 0; j < 4; ++j) {
      const int ct = wave + 8 * j;          // this wave's column tile
      const int n  = ct * 16 + m15;         // output column (= W row h)
      // B fragment: 16-bit B 32x16 layout -> lane holds bf16 c-values of W row n
      // matching A: i<8 -> k = kc + hi*8 + i, i>=8 -> k = kc + 16 + hi*8 + (i-8): two b128 loads.
      const unsigned short* wp = Wb + (size_t)n * DIM + kc + hi * 8;
      __builtin_prefetch((const void*)(wp + 64), 0, 0);   // global_prefetch_b8
      Frag16 b;
      b.h[0] = *(const u16x8*)(wp);
      b.h[1] = *(const u16x8*)(wp + 16);
#pragma unroll
      for (int rt = 0; rt < 4; ++rt) {
        acc[j * 4 + rt] = __builtin_amdgcn_wmma_f32_16x16x32_bf16(
            false, a[rt].v, false, b.v, (short)0, acc[j * 4 + rt], false, false);
      }
    }
  }

  // Epilogue: bias + relu + bf16 store per C/D layout:
  // lane holds col N = lane&15; VGPR e -> row M = e + 8*(lane>=16).
#pragma unroll
  for (int j = 0; j < 4; ++j) {
    const int ct  = wave + 8 * j;
    const int col = ct * 16 + m15;
    const float bv = bias[col];
#pragma unroll
    for (int rt = 0; rt < 4; ++rt) {
      v8f c = acc[j * 4 + rt];
#pragma unroll
      for (int e = 0; e < 8; ++e) {
        const int row = rt * 16 + hi * 8 + e;
        float r = c[e] + bv;
        r = r > 0.f ? r : 0.f;
        dstLds[row * DIM + col] = f2bf(r);
      }
    }
  }
}

__global__ void __launch_bounds__(256)
traj_heads_kernel(const float* __restrict__ X,
                  const unsigned short* __restrict__ w1b, const float* __restrict__ b1,
                  const unsigned short* __restrict__ w2b, const float* __restrict__ b2,
                  const float* __restrict__ w3, const float* __restrict__ b3,
                  float* __restrict__ out_time, float* __restrict__ packed,
                  float* __restrict__ ws2d)
{
  extern __shared__ unsigned short lds[];
  unsigned short* ldsA = lds;                       // X tile, later h2
  unsigned short* ldsB = lds + TILE_ROWS * DIM;     // h1

  const int tid   = threadIdx.x;
  const int lane  = tid & 31;   // wave32
  const int wave  = tid >> 5;   // 8 waves
  const int head  = blockIdx.y;
  const int kbase = blockIdx.x * TILE_ROWS;

  // Stage X tile fp32 -> bf16 LDS (hardware cvt)
  const float* Xt = X + (size_t)kbase * DIM;
  for (int i = tid; i < TILE_ROWS * DIM; i += 256)
    ldsA[i] = f2bf(Xt[i]);
  __syncthreads();

  fused_layer(ldsA, ldsB, w1b + (size_t)head * DIM * DIM, b1 + head * DIM, wave, lane);
  __syncthreads();
  fused_layer(ldsB, ldsA, w2b + (size_t)head * DIM * DIM, b2 + head * DIM, wave, lane);
  __syncthreads();

  // Head-specific projection (tiny N): VALU dot products out of LDS.
  int ncols, r0;
  if (head == 0)      { ncols = 1;  r0 = 0;  }
  else if (head == 1) { ncols = 6;  r0 = 1;  }
  else if (head == 2) { ncols = 4;  r0 = 7;  }
  else                { ncols = 12; r0 = 11; }

  for (int p = tid; p < TILE_ROWS * ncols; p += 256) {
    const int row = p / ncols;
    const int j   = p - row * ncols;
    const unsigned short* h2r = ldsA + row * DIM;
    const float* w3r = w3 + (size_t)(r0 + j) * DIM;
    float s = 0.f;
#pragma unroll 8
    for (int c = 0; c < DIM; ++c) s += bf2f(h2r[c]) * w3r[c];
    s += b3[r0 + j];
    const int k = kbase + row;
    if (head == 0)      out_time[k] = s;
    else if (head == 1) packed[(size_t)k * 18 + (j < 3 ? j : j + 6)] = s;
    else if (head == 2) ws2d[(size_t)k * 4 + j] = s;
    else                packed[(size_t)k * 18 + (j < 6 ? j + 3 : j + 6)] = s;
  }
}

// ---------------- loss ----------------

__device__ __forceinline__ float huber1(float d, float beta) {
  return (d < beta) ? 0.5f * d * d / beta : d - 0.5f * beta;
}

__device__ __forceinline__ void safenorm3(float v[3]) {
  float n = sqrtf(v[0] * v[0] + v[1] * v[1] + v[2] * v[2]);
  if (n > 1e-6f) { float inv = 1.f / n; v[0] *= inv; v[1] *= inv; v[2] *= inv; }
  else { v[0] = 1.f; v[1] = 0.f; v[2] = 0.f; }
}

__device__ __forceinline__ void rot6mat(const float* x, float R[3][3]) {
  float a1[3] = {x[0], x[1], x[2]};
  safenorm3(a1);
  float a2[3] = {x[3], x[4], x[5]};
  float d = a1[0] * a2[0] + a1[1] * a2[1] + a1[2] * a2[2];
  float b2[3] = {a2[0] - d * a1[0], a2[1] - d * a1[1], a2[2] - d * a1[2]};
  safenorm3(b2);
  float a3[3] = {a1[1] * b2[2] - a1[2] * b2[1],
                 a1[2] * b2[0] - a1[0] * b2[2],
                 a1[0] * b2[1] - a1[1] * b2[0]};
  safenorm3(a3);
  float bb[3] = {a3[1] * a1[2] - a3[2] * a1[1],
                 a3[2] * a1[0] - a3[0] * a1[2],
                 a3[0] * a1[1] - a3[1] * a1[0]};
  safenorm3(bb);
  R[0][0] = a1[0]; R[0][1] = a1[1]; R[0][2] = a1[2];
  R[1][0] = bb[0]; R[1][1] = bb[1]; R[1][2] = bb[2];
  R[2][0] = a3[0]; R[2][1] = a3[1]; R[2][2] = a3[2];
}

__device__ __forceinline__ float geoangle(const float Rp[3][3], const float Rg[3][3]) {
  float M[3][3];
#pragma unroll
  for (int i = 0; i < 3; ++i)
#pragma unroll
    for (int l = 0; l < 3; ++l)
      M[i][l] = Rg[0][i] * Rp[0][l] + Rg[1][i] * Rp[1][l] + Rg[2][i] * Rp[2][l];
  float tr = M[0][0] + M[1][1] + M[2][2];
  float c = fminf(fmaxf((tr - 1.f) * 0.5f, -1.f + 1e-6f), 1.f - 1e-6f);
  float v0 = M[2][1] - M[1][2];
  float v1 = M[0][2] - M[2][0];
  float v2 = M[1][0] - M[0][1];
  float s = 0.5f * sqrtf(v0 * v0 + v1 * v1 + v2 * v2) + 1e-8f;
  return atan2f(s, c);
}

__global__ void __launch_bounds__(256)
traj_loss_kernel(const float* __restrict__ gt,
                 const float* __restrict__ out_time,
                 const float* __restrict__ packed,
                 const float* __restrict__ ws2d,
                 float* __restrict__ gacc)
{
  __shared__ float sacc[10];
  if (threadIdx.x < 10) sacc[threadIdx.x] = 0.f;
  __syncthreads();

  const int k = blockIdx.x * blockDim.x + threadIdx.x;
  float v[10];
#pragma unroll
  for (int i = 0; i < 10; ++i) v[i] = 0.f;

  if (k < KROWS) {
    const float* g = gt + (size_t)k * 23;
    const float pt = out_time[k];

    const float tg = g[0];
    const float tv = (tg != -1000.f) ? 1.f : 0.f;
    const float dt = fabsf(pt - tg);
    v[0] = huber1(dt, 3.f) * tv;
    v[1] = tv;
    const float wt = __expf(-0.5f * (dt / 3.f) * (dt / 3.f)) * tv;

#pragma unroll
    for (int j = 0; j < 6; ++j) {
      const float pc = packed[(size_t)k * 18 + (j < 3 ? j : j + 6)];
      const float xg = g[1 + j];
      const float m = ((xg != -1000.f) ? 1.f : 0.f) * wt;
      v[2] += huber1(fabsf(pc - xg), 0.07f) * m;
      v[3] += m;
    }
#pragma unroll
    for (int j = 0; j < 4; ++j) {
      const float pu = ws2d[(size_t)k * 4 + j] * (1.f / 1408.f);
      const float ug = g[7 + j];
      const float m = ((ug != -1000.f) ? 1.f : 0.f) * wt;
      v[4] += huber1(fabsf(pu - ug * (1.f / 1408.f)), 0.02f) * m;
      v[5] += m;
    }

    float pq[12], qg[12];
#pragma unroll
    for (int j = 0; j < 12; ++j) {
      pq[j] = packed[(size_t)k * 18 + (j < 6 ? j + 3 : j + 6)];
      qg[j] = g[11 + j];
      const float m = ((qg[j] != -1000.f) ? 1.f : 0.f) * wt;
      v[6] += huber1(fabsf(pq[j] - qg[j]), 0.2f) * m;
      v[7] += m;
    }

    bool Lv = true, Rv = true;
#pragma unroll
    for (int j = 0; j < 6; ++j)  Lv = Lv && (qg[j] != -1000.f);
#pragma unroll
    for (int j = 6; j < 12; ++j) Rv = Rv && (qg[j] != -1000.f);

    const float safe6[6] = {1.f, 0.f, 0.f, 0.f, 1.f, 0.f};
    float Lt[6], Rt[6];
#pragma unroll
    for (int j = 0; j < 6; ++j) {
      Lt[j] = Lv ? qg[j] : safe6[j];
      Rt[j] = Rv ? qg[6 + j] : safe6[j];
    }

    float Rp[3][3], Rg[3][3];
    rot6mat(pq, Rp);     rot6mat(Lt, Rg);
    const float angL = geoangle(Rp, Rg);
    rot6mat(pq + 6, Rp); rot6mat(Rt, Rg);
    const float angR = geoangle(Rp, Rg);

    v[8] = (Lv ? angL : 0.f) + (Rv ? angR : 0.f);
    v[9] = (Lv ? 1.f : 0.f) + (Rv ? 1.f : 0.f);
  }

#pragma unroll
  for (int i = 0; i < 10; ++i) atomicAdd(&sacc[i], v[i]);  // ds_add_f32
  __syncthreads();
  if (threadIdx.x < 10) atomicAdd(&gacc[threadIdx.x], sacc[threadIdx.x]);
}

__global__ void zero_acc_kernel(float* gacc) {
  if (threadIdx.x < 10) gacc[threadIdx.x] = 0.f;
}

__global__ void finalize_kernel(const float* __restrict__ gacc, float* __restrict__ out_loss) {
  const float time_loss = gacc[0] / fmaxf(gacc[1], 1.f);
  const float l3 = gacc[2] / fmaxf(gacc[3], 1.f);
  const float l2 = gacc[4] / fmaxf(gacc[5], 1.f);
  const float lq = gacc[6] / fmaxf(gacc[7], 1.f);
  const float lr = 0.15f * (gacc[8] / fmaxf(gacc[9], 1e-8f));
  *out_loss = 1.f * time_loss + 2.f * l3 + 0.5f * l2 + 0.5f * lq + lr;
}

extern "C" void kernel_launch(void* const* d_in, const int* in_sizes, int n_in,
                              void* d_out, int out_size, void* d_ws, size_t ws_size,
                              hipStream_t stream) {
  const float* X  = (const float*)d_in[0];
  const float* gt = (const float*)d_in[1];
  const float* w1 = (const float*)d_in[2];
  const float* b1 = (const float*)d_in[3];
  const float* w2 = (const float*)d_in[4];
  const float* b2 = (const float*)d_in[5];
  const float* w3 = (const float*)d_in[6];
  const float* b3 = (const float*)d_in[7];

  float* out_time = (float*)d_out;                  // K
  float* packed   = out_time + KROWS;               // K*18
  float* out_loss = packed + (size_t)KROWS * 18;    // 1

  // Workspace layout (aligned): pred_2d | accumulators(16) | bf16 w1 | bf16 w2
  float* ws2d = (float*)d_ws;                              // K*4 floats
  float* gacc = ws2d + (size_t)KROWS * 4;                  // 10 (pad 16)
  unsigned short* w1b = (unsigned short*)(gacc + 16);      // 4*512*512 bf16
  unsigned short* w2b = w1b + (size_t)4 * DIM * DIM;       // 4*512*512 bf16

  const int wN4 = (4 * DIM * DIM) / 4;                     // float4 count
  cvt_w_kernel<<<(wN4 + 255) / 256, 256, 0, stream>>>(w1, w1b, wN4);
  cvt_w_kernel<<<(wN4 + 255) / 256, 256, 0, stream>>>(w2, w2b, wN4);

  zero_acc_kernel<<<1, 32, 0, stream>>>(gacc);

  const size_t shmem = (size_t)TILE_ROWS * DIM * 2 * sizeof(unsigned short); // 128 KB
  (void)hipFuncSetAttribute((const void*)traj_heads_kernel,
                            hipFuncAttributeMaxDynamicSharedMemorySize, (int)shmem);

  dim3 grid(KROWS / TILE_ROWS, 4);
  traj_heads_kernel<<<grid, 256, shmem, stream>>>(X, w1b, b1, w2b, b2, w3, b3,
                                                  out_time, packed, ws2d);

  traj_loss_kernel<<<KROWS / 256, 256, 0, stream>>>(gt, out_time, packed, ws2d, gacc);
  finalize_kernel<<<1, 1, 0, stream>>>(gacc, out_loss);
}
